// HeteroGraphSAGE_63651415327099
// MI455X (gfx1250) — matrix operand order repaired and checked by reading, not compiled, not run on hardware
//
#include <hip/hip_runtime.h>
#include <stdint.h>

// ---------------------------------------------------------------------------
// HeteroGraphSAGE forward for MI455X (gfx1250, wave32).
//  - Dense GEMMs on the WMMA path: v_wmma_f32_16x16x32_bf16 (bf16 in, f32 acc).
//  - Weights staged in LDS in *fragment-major* bf16 layout so each lane loads
//    a full 16-element B fragment with one 32B vector LDS load.
//  - Native packed f32->bf16 converts for A fragments.
//  - Edge mean-aggregation via L2-resident global_atomic_add_f32 scatter.
//  - Degree normalization fused into the GEMM A-operand load.
//  - LayerNorm+ReLU fused, one wave32 per 128-wide row.
// ---------------------------------------------------------------------------

#define N_NODES 50000
#define N_EDGES 500000
#define D_HID   128

typedef __attribute__((ext_vector_type(16))) __bf16 v16bf;
typedef __attribute__((ext_vector_type(16))) short  v16s;
typedef __attribute__((ext_vector_type(2)))  __bf16 v2bf;
typedef __attribute__((ext_vector_type(8)))  float  v8f;

__device__ __forceinline__ v2bf cvt2(float a, float b) {
#if __has_builtin(__builtin_amdgcn_cvt_pk_bf16_f32)
  return __builtin_amdgcn_cvt_pk_bf16_f32(a, b);
#else
  v2bf r; r[0] = (__bf16)a; r[1] = (__bf16)b; return r;
#endif
}

// A fragment (16x32 bf16, MxK). ISA layout: lane L holds row M=L%16;
// lanes 0-15: K = kc*32 + {0..7, 16..23}; lanes 16-31: +8.
template<int DIN>
__device__ __forceinline__ v16bf load_a_frag(const float* __restrict__ h, int row,
                                             int kc, int half, float scale) {
  const float* p = h + (size_t)row * DIN + kc * 32 + half * 8;
  v16bf a;
#pragma unroll
  for (int i = 0; i < 4; ++i) {
    v2bf lo = cvt2(p[2 * i] * scale,      p[2 * i + 1] * scale);
    v2bf hi = cvt2(p[16 + 2 * i] * scale, p[16 + 2 * i + 1] * scale);
    a[2 * i]     = lo[0]; a[2 * i + 1]     = lo[1];
    a[8 + 2 * i] = hi[0]; a[8 + 2 * i + 1] = hi[1];
  }
  return a;
}

// B fragment from fragment-major LDS: slot [(kc*NT+nt)*32 + lane] holds the
// lane's 16 bf16 values contiguously -> one 32B LDS vector load.
template<int NT>
__device__ __forceinline__ v16bf load_b_frag(const v16s* __restrict__ w,
                                             int kc, int nt, int lane) {
  return __builtin_bit_cast(v16bf, w[(kc * NT + nt) * 32 + lane]);
}

// Fused SAGE GEMM:  out = Hs @ Ws + bias [+ (Hn/deg) @ Wn]
// Block: 256 threads = 8 waves; each wave owns one 16-row M-tile and all
// DOUT/16 N-tiles. Weights staged in LDS (bf16, fragment-major) per block.
// Safe to alias out==Hn: each row is read (A frags, up front) and written
// (stores, at the end) by exactly one wave.
template<int DIN, int DOUT, bool NEIGH>
__global__ void __launch_bounds__(256)
sage_gemm(const float* __restrict__ Hs, const float* __restrict__ Hn,
          const float* __restrict__ deg,
          const float* __restrict__ Ws, const float* __restrict__ Wn,
          const float* __restrict__ bias,
          float* __restrict__ out, int mtiles) {
  constexpr int KC = DIN / 32;
  constexpr int NT = DOUT / 16;
  constexpr int NFRAG = KC * NT * 32;          // (kc,nt) fragments x 32 lanes
  __shared__ v16s lws[NFRAG];
  __shared__ v16s lwn[NEIGH ? NFRAG : 1];

  // Stage weights W[K][N] (f32, row-major) -> bf16 fragment-major layout.
  // i = ((kc*NT + nt)*32 + lane)*16 + j ;  B element (K,N):
  //   N = nt*16 + lane%16 ; K = kc*32 + (lane/16)*8 + (j%8) + (j/8)*16
  {
    __bf16* ps = (__bf16*)lws;
    __bf16* pn = (__bf16*)lwn;
    for (int i = threadIdx.x; i < DIN * DOUT; i += 256) {
      int j    = i & 15;
      int lane = (i >> 4) & 31;
      int f    = i >> 9;
      int nt   = f % NT;
      int kc   = f / NT;
      int col  = nt * 16 + (lane & 15);
      int K    = kc * 32 + (lane >> 4) * 8 + (j & 7) + (j >> 3) * 16;
      ps[i] = (__bf16)Ws[K * DOUT + col];
      if (NEIGH) pn[i] = (__bf16)Wn[K * DOUT + col];
    }
  }
  __syncthreads();

  const int wave = threadIdx.x >> 5;
  const int lane = threadIdx.x & 31;
  const int mt   = blockIdx.x * 8 + wave;
  if (mt >= mtiles) return;

  const int half = lane >> 4;
  const int row  = mt * 16 + (lane & 15);

  v16bf aS[KC];
  v16bf aN[NEIGH ? KC : 1];
#pragma unroll
  for (int kc = 0; kc < KC; ++kc) aS[kc] = load_a_frag<DIN>(Hs, row, kc, half, 1.0f);
  if (NEIGH) {
    const float invd = 1.0f / fmaxf(deg[row], 1.0f);   // mean-agg normalization
#pragma unroll
    for (int kc = 0; kc < KC; ++kc) aN[kc] = load_a_frag<DIN>(Hn, row, kc, half, invd);
  }

#pragma unroll
  for (int nt = 0; nt < NT; ++nt) {
    const int col = nt * 16 + (lane & 15);
    const float bv = bias[col];
    v8f c;
#pragma unroll
    for (int r = 0; r < 8; ++r) c[r] = bv;
#pragma unroll
    for (int kc = 0; kc < KC; ++kc) {
      v16bf bS = load_b_frag<NT>(lws, kc, nt, lane);
      c = __builtin_amdgcn_wmma_f32_16x16x32_bf16(false, aS[kc], false, bS,
                                                  (short)0, c, false, false);
      if (NEIGH) {
        v16bf bN = load_b_frag<NT>(lwn, kc, nt, lane);
        c = __builtin_amdgcn_wmma_f32_16x16x32_bf16(false, aN[kc], false, bN,
                                                    (short)0, c, false, false);
      }
    }
    // C/D layout: VGPR r -> row = mt*16 + r + 8*half, col = nt*16 + lane%16.
#pragma unroll
    for (int r = 0; r < 8; ++r)
      out[(size_t)(mt * 16 + half * 8 + r) * DOUT + col] = c[r];
  }
}

// in-degree via exact fp32 counting (counts < 2^24 -> deterministic value)
__global__ void deg_count(const int* __restrict__ dst, float* __restrict__ deg, int n) {
  int i = blockIdx.x * blockDim.x + threadIdx.x;
  if (i < n) atomicAdd(&deg[dst[i]], 1.0f);
}

// Edge scatter: agg[dst] += h[src], d=128. One thread per (edge, 4-float chunk).
__global__ void edge_agg(const float* __restrict__ h, const int* __restrict__ src,
                         const int* __restrict__ dst, float* __restrict__ agg, int n) {
  int t = blockIdx.x * blockDim.x + threadIdx.x;
  int e = t >> 5, c = t & 31;
  if (e >= n) return;
  int s = src[e], d = dst[e];
  float4 v = reinterpret_cast<const float4*>(h)[(size_t)s * 32 + c];
  float* o = agg + (size_t)d * D_HID + c * 4;
  atomicAdd(o + 0, v.x);
  atomicAdd(o + 1, v.y);
  atomicAdd(o + 2, v.z);
  atomicAdd(o + 3, v.w);
}

// LayerNorm + ReLU, d=128, one wave32 per row (4 values/lane).
__global__ void ln_relu(float* __restrict__ x, const float* __restrict__ g,
                        const float* __restrict__ b, int nrows) {
  int wave = threadIdx.x >> 5, lane = threadIdx.x & 31;
  int row = blockIdx.x * 8 + wave;
  if (row >= nrows) return;
  float v[4], s = 0.f, ss = 0.f;
#pragma unroll
  for (int j = 0; j < 4; ++j) {
    v[j] = x[(size_t)row * D_HID + j * 32 + lane];
    s += v[j]; ss += v[j] * v[j];
  }
#pragma unroll
  for (int off = 16; off > 0; off >>= 1) {    // wave32 reduction
    s  += __shfl_xor(s,  off, 32);
    ss += __shfl_xor(ss, off, 32);
  }
  const float mu  = s * (1.0f / D_HID);
  const float var = ss * (1.0f / D_HID) - mu * mu;
  const float r   = rsqrtf(var + 1e-5f);
#pragma unroll
  for (int j = 0; j < 4; ++j) {
    int c = j * 32 + lane;
    float y = (v[j] - mu) * r * g[c] + b[c];
    x[(size_t)row * D_HID + c] = fmaxf(y, 0.0f);
  }
}

// ---------------------------------------------------------------------------
// Input index map (setup_inputs() recursive insertion order):
// 0 x_user 1 x_item 2 eu2i_src 3 eu2i_dst 4 ei2u_src 5 ei2u_dst
// 6 proj_user.W 7 proj_user.b 8 proj_item.W 9 proj_item.b
// 10+6l: u2i{W_self,b_self,W_neigh}, i2u{W_self,b_self,W_neigh}
// 28+2l: norms[l]{g,b}
// ---------------------------------------------------------------------------
extern "C" void kernel_launch(void* const* d_in, const int* in_sizes, int n_in,
                              void* d_out, int out_size, void* d_ws, size_t ws_size,
                              hipStream_t stream) {
  const float* x_user = (const float*)d_in[0];
  const float* x_item = (const float*)d_in[1];
  const int* eu2i_src = (const int*)d_in[2];
  const int* eu2i_dst = (const int*)d_in[3];
  const int* ei2u_src = (const int*)d_in[4];
  const int* ei2u_dst = (const int*)d_in[5];
  const float* puW = (const float*)d_in[6];
  const float* puB = (const float*)d_in[7];
  const float* piW = (const float*)d_in[8];
  const float* piB = (const float*)d_in[9];

  const size_t NF = (size_t)N_NODES * D_HID;
  if (ws_size < (4 * NF + 2 * N_NODES) * sizeof(float)) return;
  float* ws = (float*)d_ws;
  float* U0 = ws;
  float* U1 = ws + NF;
  float* I0 = ws + 2 * NF;
  float* I1 = ws + 3 * NF;
  float* degU = ws + 4 * NF;
  float* degI = degU + N_NODES;

  const int MT = N_NODES / 16;           // 3125 exact M-tiles
  dim3 gG((MT + 7) / 8), bG(256);
  dim3 gE((N_EDGES + 255) / 256), b256(256);
  dim3 gA((N_EDGES * 32 + 255) / 256);
  dim3 gL((N_NODES + 7) / 8);

  // Degrees (constant across layers).
  hipMemsetAsync(degU, 0, 2 * N_NODES * sizeof(float), stream);
  deg_count<<<gE, b256, 0, stream>>>(eu2i_dst, degI, N_EDGES);
  deg_count<<<gE, b256, 0, stream>>>(ei2u_dst, degU, N_EDGES);

  // Input projections.
  sage_gemm<128, 128, false><<<gG, bG, 0, stream>>>(x_user, nullptr, nullptr,
                                                    puW, nullptr, puB, U0, MT);
  sage_gemm<64, 128, false><<<gG, bG, 0, stream>>>(x_item, nullptr, nullptr,
                                                   piW, nullptr, piB, I0, MT);

  float* curU = U0; float* curI = I0;
  float* tmpU = U1; float* tmpI = I1;

  for (int l = 0; l < 3; ++l) {
    const int base = 10 + l * 6;
    const float* u2i_Ws = (const float*)d_in[base + 0];
    const float* u2i_b  = (const float*)d_in[base + 1];
    const float* u2i_Wn = (const float*)d_in[base + 2];
    const float* i2u_Ws = (const float*)d_in[base + 3];
    const float* i2u_b  = (const float*)d_in[base + 4];
    const float* i2u_Wn = (const float*)d_in[base + 5];

    hipMemsetAsync(tmpU, 0, NF * sizeof(float), stream);
    hipMemsetAsync(tmpI, 0, NF * sizeof(float), stream);
    edge_agg<<<gA, b256, 0, stream>>>(curU, eu2i_src, eu2i_dst, tmpI, N_EDGES);
    edge_agg<<<gA, b256, 0, stream>>>(curI, ei2u_src, ei2u_dst, tmpU, N_EDGES);

    if (l < 2) {
      sage_gemm<128, 128, true><<<gG, bG, 0, stream>>>(curI, tmpI, degI,
                                                       u2i_Ws, u2i_Wn, u2i_b, tmpI, MT);
      sage_gemm<128, 128, true><<<gG, bG, 0, stream>>>(curU, tmpU, degU,
                                                       i2u_Ws, i2u_Wn, i2u_b, tmpU, MT);
      const float* ng = (const float*)d_in[28 + 2 * l];
      const float* nb = (const float*)d_in[29 + 2 * l];
      ln_relu<<<gL, b256, 0, stream>>>(tmpU, ng, nb, N_NODES);
      ln_relu<<<gL, b256, 0, stream>>>(tmpI, ng, nb, N_NODES);
      float* t;
      t = curU; curU = tmpU; tmpU = t;
      t = curI; curI = tmpI; tmpI = t;
    } else {
      float* outU = (float*)d_out;                       // hu first
      float* outI = outU + (size_t)N_NODES * 64;         // then hi
      sage_gemm<128, 64, true><<<gG, bG, 0, stream>>>(curI, tmpI, degI,
                                                      u2i_Ws, u2i_Wn, u2i_b, outI, MT);
      sage_gemm<128, 64, true><<<gG, bG, 0, stream>>>(curU, tmpU, degU,
                                                      i2u_Ws, i2u_Wn, i2u_b, outU, MT);
    }
  }
}
